// GlmDSAMoE_62895501082721
// MI455X (gfx1250) — matrix operand order repaired and checked by reading, not compiled, not run on hardware
//
#include <hip/hip_runtime.h>
#include <hip/hip_bf16.h>

typedef unsigned int u32;
typedef unsigned short u16;
typedef __attribute__((ext_vector_type(16))) __bf16 bf16x16;
typedef __attribute__((ext_vector_type(4)))  __bf16 bf16x4;
typedef __attribute__((ext_vector_type(8)))  float   f32x8;
typedef __attribute__((ext_vector_type(4)))  float   f32x4;
typedef __attribute__((ext_vector_type(4)))  u32     u32x4;
typedef __attribute__((ext_vector_type(8)))  int     i32x8;
typedef __attribute__((ext_vector_type(4)))  int     i32x4;

#define T_TOK 1024
#define DMODEL 2048
#define IDIM   1024
#define NEXP   32
#define TOPK   8
#define NGRP   8
#define TOPG   4
#define SCALING 2.5f
#define ROUTED_ROWS (T_TOK * TOPK)        // 8192
#define TOTAL_ROWS  (ROUTED_ROWS + T_TOK) // 9216

union FragAB { bf16x16 v; u32x4 q[2]; u16 s[16]; };

__device__ __forceinline__ void store_bf4(u16* dst, f32x4 v) {
    bf16x4 b = __builtin_convertvector(v, bf16x4);   // v_cvt_pk_bf16_f32 x2
    *(bf16x4*)dst = b;                               // 8-byte store
}
__device__ __forceinline__ u16 f2bf1(float f) {
    union { __bf16 b; u16 u; } c; c.b = (__bf16)f; return c.u;
}

// ---------------- zero output + counts ----------------
__global__ void moe_zero_kernel(float* __restrict__ out, int n, int* __restrict__ counts) {
    int i = blockIdx.x * blockDim.x + threadIdx.x;
    if (i < n) out[i] = 0.0f;
    if (i < NEXP) counts[i] = 0;
}

// ---------------- gate logits -> sigmoid scores [T,E] ----------------
__global__ __launch_bounds__(256) void moe_gate_kernel(
    const float* __restrict__ x, const float* __restrict__ gw,
    float* __restrict__ scores) {
    int t = blockIdx.x;
    __shared__ float xs[DMODEL];
    __shared__ float part[256];
    for (int i = threadIdx.x; i < DMODEL; i += 256) xs[i] = x[(size_t)t * DMODEL + i];
    __syncthreads();
    int e   = threadIdx.x >> 3;   // 0..31
    int sub = threadIdx.x & 7;    // 0..7
    const float* w  = gw + (size_t)e * DMODEL + sub * 256;
    const float* xv = xs + sub * 256;
    float acc = 0.f;
    #pragma unroll 8
    for (int i = 0; i < 256; ++i) acc += xv[i] * w[i];
    part[threadIdx.x] = acc;
    __syncthreads();
    if (sub == 0) {
        float s = 0.f;
        #pragma unroll
        for (int i = 0; i < 8; ++i) s += part[(e << 3) + i];
        scores[t * NEXP + e] = __fdividef(1.0f, 1.0f + __expf(-s));
    }
}

// ---------------- routing: noaux_tc group top-k ----------------
__global__ void moe_route_kernel(
    const float* __restrict__ scores, const float* __restrict__ e_bias,
    int* __restrict__ counts, int* __restrict__ staged_id, float* __restrict__ staged_wt) {
    int t = blockIdx.x * blockDim.x + threadIdx.x;
    if (t >= T_TOK) return;
    float sc[NEXP], sfc[NEXP];
    #pragma unroll
    for (int e = 0; e < NEXP; ++e) {
        sc[e]  = scores[t * NEXP + e];
        sfc[e] = sc[e] + e_bias[e];
    }
    float gs[NGRP];
    #pragma unroll
    for (int g = 0; g < NGRP; ++g) {
        float m1 = -1e30f, m2 = -1e30f;
        #pragma unroll
        for (int j = 0; j < 4; ++j) {
            float v = sfc[g * 4 + j];
            if (v > m1) { m2 = m1; m1 = v; } else if (v > m2) { m2 = v; }
        }
        gs[g] = m1 + m2;
    }
    u32 gmask = 0;
    #pragma unroll
    for (int k = 0; k < TOPG; ++k) {
        float best = -1e30f; int bi = 0;
        #pragma unroll
        for (int g = 0; g < NGRP; ++g)
            if (!((gmask >> g) & 1u) && gs[g] > best) { best = gs[g]; bi = g; }
        gmask |= (1u << bi);
    }
    float masked[NEXP];
    #pragma unroll
    for (int e = 0; e < NEXP; ++e)
        masked[e] = ((gmask >> (e >> 2)) & 1u) ? sfc[e] : 0.0f;
    int tidx[TOPK]; float sum = 0.f; float tw[TOPK];
    #pragma unroll
    for (int k = 0; k < TOPK; ++k) {
        float best = -1e30f; int bi = 0;
        #pragma unroll
        for (int e = 0; e < NEXP; ++e)
            if (masked[e] > best) { best = masked[e]; bi = e; }
        masked[bi] = -1e30f;
        tidx[k] = bi; tw[k] = sc[bi]; sum += sc[bi];
    }
    float inv = SCALING * __fdividef(1.0f, sum + 1e-20f);
    #pragma unroll
    for (int k = 0; k < TOPK; ++k) {
        int e   = tidx[k];
        int pos = atomicAdd(&counts[e], 1);
        staged_id[e * T_TOK + pos] = t;
        staged_wt[e * T_TOK + pos] = tw[k] * inv;
    }
}

// ---------------- tiny exclusive scan over expert counts ----------------
__global__ void moe_scan_kernel(const int* __restrict__ counts, int* __restrict__ offsets) {
    if (threadIdx.x == 0) {
        int s = 0;
        for (int e = 0; e < NEXP; ++e) { offsets[e] = s; s += counts[e]; }
        offsets[NEXP] = s;   // == 8192
    }
}

// ---------------- FFN stage 1: h = silu(x w1^T) * (x w3^T) ------------------
// block tile M=128 x N=64 over I; 8 waves as 4(M) x 2(N); each wave 32x32
// grid: (1024/128=8, IDIM/64=16, 33)
__global__ __launch_bounds__(256) void moe_ffn1_kernel(
    const float* __restrict__ x,
    const float* __restrict__ w1, const float* __restrict__ w3,
    const float* __restrict__ sw1, const float* __restrict__ sw3,
    const int* __restrict__ counts, const int* __restrict__ offsets,
    const int* __restrict__ staged_id, u16* __restrict__ h) {
    const int e   = blockIdx.z;
    const int cnt = (e == NEXP) ? T_TOK : counts[e];
    const int m0  = blockIdx.x * 128;
    if (m0 >= cnt) return;
    const int n0  = blockIdx.y * 64;
    const int rowbase = (e == NEXP) ? ROUTED_ROWS : offsets[e];
    const float* W1 = (e == NEXP) ? sw1 : (w1 + (size_t)e * IDIM * DMODEL);
    const float* W3 = (e == NEXP) ? sw3 : (w3 + (size_t)e * IDIM * DMODEL);

    __shared__ u16 As[128 * 32];
    __shared__ u16 B1s[64 * 32];
    __shared__ u16 B3s[64 * 32];
    __shared__ int tok[128];

    const int tid  = threadIdx.x;
    const int lane = tid & 31;
    const int wave = tid >> 5;
    const int wm = wave >> 1;   // 0..3 (M subtile of 32)
    const int wn = wave & 1;    // 0..1 (N subtile of 32)

    if (tid < 128) {
        int pos = m0 + tid;
        tok[tid] = (pos < cnt) ? ((e == NEXP) ? pos : staged_id[e * T_TOK + pos]) : -1;
    }

    f32x8 accG[2][2] = {}; f32x8 accU[2][2] = {};

    const int rA = tid >> 1;          // 0..127
    const int kA = (tid & 1) * 16;    // 0 or 16
    const int rB = tid >> 2;          // 0..63
    const int kB = (tid & 3) * 8;     // 0,8,16,24

    for (int k0 = 0; k0 < DMODEL; k0 += 32) {
        __syncthreads();
        // --- A tile: gathered token rows, fp32 -> bf16 (packed cvt) ---
        {
            int tk = tok[rA];
            f32x4 v0 = {}, v1 = {}, v2 = {}, v3 = {};
            if (tk >= 0) {
                const float* px = x + (size_t)tk * DMODEL + k0 + kA;
                v0 = *(const f32x4*)px;       v1 = *(const f32x4*)(px + 4);
                v2 = *(const f32x4*)(px + 8); v3 = *(const f32x4*)(px + 12);
            }
            u16* d = &As[rA * 32 + kA];
            store_bf4(d, v0); store_bf4(d + 4, v1);
            store_bf4(d + 8, v2); store_bf4(d + 12, v3);
        }
        // --- B tiles: w1 / w3 rows, fp32 -> bf16 ---
        {
            const float* p1 = W1 + (size_t)(n0 + rB) * DMODEL + k0 + kB;
            const float* p3 = W3 + (size_t)(n0 + rB) * DMODEL + k0 + kB;
            store_bf4(&B1s[rB * 32 + kB],     *(const f32x4*)p1);
            store_bf4(&B1s[rB * 32 + kB + 4], *(const f32x4*)(p1 + 4));
            store_bf4(&B3s[rB * 32 + kB],     *(const f32x4*)p3);
            store_bf4(&B3s[rB * 32 + kB + 4], *(const f32x4*)(p3 + 4));
        }
        __syncthreads();
        // --- fragments + 8 WMMAs per wave ---
        FragAB a[2], b1[2], b3[2];
        const int khalf = (lane >> 4) * 8;
        const int kb    = (lane >> 4) * 8;
        #pragma unroll
        for (int sm = 0; sm < 2; ++sm) {
            int mrow = wm * 32 + sm * 16 + (lane & 15);
            a[sm].q[0] = *(const u32x4*)&As[mrow * 32 + khalf];
            a[sm].q[1] = *(const u32x4*)&As[mrow * 32 + khalf + 16];
        }
        #pragma unroll
        for (int sn = 0; sn < 2; ++sn) {
            int ncol = wn * 32 + sn * 16 + (lane & 15);
            b1[sn].q[0] = *(const u32x4*)&B1s[ncol * 32 + kb];
            b1[sn].q[1] = *(const u32x4*)&B1s[ncol * 32 + kb + 16];
            b3[sn].q[0] = *(const u32x4*)&B3s[ncol * 32 + kb];
            b3[sn].q[1] = *(const u32x4*)&B3s[ncol * 32 + kb + 16];
        }
        #pragma unroll
        for (int sm = 0; sm < 2; ++sm)
            #pragma unroll
            for (int sn = 0; sn < 2; ++sn) {
                accG[sm][sn] = __builtin_amdgcn_wmma_f32_16x16x32_bf16(
                    false, a[sm].v, false, b1[sn].v, (short)0, accG[sm][sn], false, false);
                accU[sm][sn] = __builtin_amdgcn_wmma_f32_16x16x32_bf16(
                    false, a[sm].v, false, b3[sn].v, (short)0, accU[sm][sn], false, false);
            }
    }
    // --- epilogue: silu(g)*u -> bf16 h ---
    const int N   = lane & 15;
    const int mhi = (lane >> 4) * 8;
    #pragma unroll
    for (int sm = 0; sm < 2; ++sm)
        #pragma unroll
        for (int sn = 0; sn < 2; ++sn)
            #pragma unroll
            for (int j = 0; j < 8; ++j) {
                int pos = m0 + wm * 32 + sm * 16 + j + mhi;
                if (pos < cnt) {
                    float g = accG[sm][sn][j], u = accU[sm][sn][j];
                    float hv = g * __fdividef(1.0f, 1.0f + __expf(-g)) * u;
                    h[(size_t)(rowbase + pos) * IDIM + n0 + wn * 32 + sn * 16 + N] = f2bf1(hv);
                }
            }
}

// ---------------- FFN stage 2: out += wt * (h w2^T) -------------------------
// A tiles (bf16 h) fetched with the Tensor Data Mover; OOB rows zero-filled by
// hardware via tensor_dim1 = cnt - m0.  grid: (8, DMODEL/64=32, 33)
__global__ __launch_bounds__(256) void moe_ffn2_kernel(
    const u16* __restrict__ h,
    const float* __restrict__ w2, const float* __restrict__ sw2,
    const int* __restrict__ counts, const int* __restrict__ offsets,
    const int* __restrict__ staged_id, const float* __restrict__ staged_wt,
    float* __restrict__ out) {
    const int e   = blockIdx.z;
    const int cnt = (e == NEXP) ? T_TOK : counts[e];
    const int m0  = blockIdx.x * 128;
    if (m0 >= cnt) return;
    const int n0  = blockIdx.y * 64;   // over D
    const int rowbase = (e == NEXP) ? ROUTED_ROWS : offsets[e];
    const float* W2 = (e == NEXP) ? sw2 : (w2 + (size_t)e * DMODEL * IDIM);

    __shared__ u16 As[128 * 32];
    __shared__ u16 Bs[64 * 32];
    __shared__ int   tok[128];
    __shared__ float wt[128];

    const int tid  = threadIdx.x;
    const int lane = tid & 31;
    const int wave = tid >> 5;
    const int wm = wave >> 1;
    const int wn = wave & 1;

    if (tid < 128) {
        int pos = m0 + tid;
        if (pos < cnt) {
            tok[tid] = (e == NEXP) ? pos : staged_id[e * T_TOK + pos];
            wt[tid]  = (e == NEXP) ? 1.0f : staged_wt[e * T_TOK + pos];
        } else { tok[tid] = 0; wt[tid] = 0.0f; }
    }

    // uniform pieces of the TDM descriptor
    const u32 lds_a   = (u32)(uintptr_t)&As[0];
    const u32 cnt_rem = (u32)(cnt - m0);          // >= 1
    const size_t grow = ((size_t)(rowbase + m0) * IDIM) * sizeof(u16);

    f32x8 acc[2][2] = {};
    const int rB = tid >> 2;
    const int kB = (tid & 3) * 8;

    for (int k0 = 0; k0 < IDIM; k0 += 32) {
        __syncthreads();
        // --- A tile via Tensor Data Mover: 128 rows x 32 bf16, row stride I ---
        if (wave == 0) {
            size_t gaddr = (size_t)(uintptr_t)h + grow + (size_t)k0 * sizeof(u16);
            u32x4 g0;
            g0[0] = 1u;                                       // count=1, user D#
            g0[1] = lds_a;                                    // lds_addr
            g0[2] = (u32)(gaddr & 0xFFFFFFFFu);               // global_addr[31:0]
            g0[3] = (u32)((gaddr >> 32) & 0x1FFFFFFu) | (2u << 30); // addr[56:32] | type=2
            i32x8 g1;
            g1[0] = (int)(1u << 16);                          // data_size = 2 bytes
            g1[1] = (int)((u32)IDIM << 16);                   // tensor_dim0[15:0] @ [63:48]
            g1[2] = (int)((cnt_rem & 0xFFFFu) << 16);         // dim0 hi=0 | tensor_dim1 lo
            g1[3] = (int)((32u << 16) | ((cnt_rem >> 16) & 0xFFFFu)); // dim1 hi | tile_dim0=32
            g1[4] = (int)128u;                                // tile_dim1=128, tile_dim2=0
            g1[5] = (int)IDIM;                                // tensor_dim0_stride[31:0]
            g1[6] = 0;                                        // stride hi, dim1_stride lo
            g1[7] = 0;
            i32x4 gz4 = {0, 0, 0, 0};
            i32x8 gz8 = {0, 0, 0, 0, 0, 0, 0, 0};
            __builtin_amdgcn_tensor_load_to_lds(g0, g1, gz4, gz4, gz8, 0);
        }
        // --- B tile: w2 rows (d-major, contiguous i), fp32 -> bf16 ---
        {
            const float* p = W2 + (size_t)(n0 + rB) * IDIM + k0 + kB;
            store_bf4(&Bs[rB * 32 + kB],     *(const f32x4*)p);
            store_bf4(&Bs[rB * 32 + kB + 4], *(const f32x4*)(p + 4));
        }
        if (wave == 0) __builtin_amdgcn_s_wait_tensorcnt(0);
        __syncthreads();

        FragAB a[2], b[2];
        const int khalf = (lane >> 4) * 8;
        const int kb    = (lane >> 4) * 8;
        #pragma unroll
        for (int sm = 0; sm < 2; ++sm) {
            int mrow = wm * 32 + sm * 16 + (lane & 15);
            a[sm].q[0] = *(const u32x4*)&As[mrow * 32 + khalf];
            a[sm].q[1] = *(const u32x4*)&As[mrow * 32 + khalf + 16];
        }
        #pragma unroll
        for (int sn = 0; sn < 2; ++sn) {
            int ncol = wn * 32 + sn * 16 + (lane & 15);
            b[sn].q[0] = *(const u32x4*)&Bs[ncol * 32 + kb];
            b[sn].q[1] = *(const u32x4*)&Bs[ncol * 32 + kb + 16];
        }
        #pragma unroll
        for (int sm = 0; sm < 2; ++sm)
            #pragma unroll
            for (int sn = 0; sn < 2; ++sn)
                acc[sm][sn] = __builtin_amdgcn_wmma_f32_16x16x32_bf16(
                    false, a[sm].v, false, b[sn].v, (short)0, acc[sm][sn], false, false);
    }
    // --- epilogue: scale by routing weight, combine into output ---
    const int N   = lane & 15;
    const int mhi = (lane >> 4) * 8;
    #pragma unroll
    for (int sm = 0; sm < 2; ++sm)
        #pragma unroll
        for (int sn = 0; sn < 2; ++sn)
            #pragma unroll
            for (int j = 0; j < 8; ++j) {
                int M   = wm * 32 + sm * 16 + j + mhi;
                int pos = m0 + M;
                if (pos < cnt) {
                    atomicAdd(&out[(size_t)tok[M] * DMODEL + n0 + wn * 32 + sn * 16 + N],
                              acc[sm][sn][j] * wt[M]);
                }
            }
}

extern "C" void kernel_launch(void* const* d_in, const int* in_sizes, int n_in,
                              void* d_out, int out_size, void* d_ws, size_t ws_size,
                              hipStream_t stream) {
    const float* x      = (const float*)d_in[0];
    const float* gw     = (const float*)d_in[1];
    const float* e_bias = (const float*)d_in[2];
    const float* w1     = (const float*)d_in[3];
    const float* w2     = (const float*)d_in[4];
    const float* w3     = (const float*)d_in[5];
    const float* sw1    = (const float*)d_in[6];
    const float* sw2    = (const float*)d_in[7];
    const float* sw3    = (const float*)d_in[8];
    float* out = (float*)d_out;

    char* p = (char*)d_ws;
    float* scores    = (float*)p;             p += (size_t)T_TOK * NEXP * 4;
    int*   counts    = (int*)p;               p += 64 * 4;
    int*   offsets   = (int*)p;               p += 64 * 4;
    int*   staged_id = (int*)p;               p += (size_t)NEXP * T_TOK * 4;
    float* staged_wt = (float*)p;             p += (size_t)NEXP * T_TOK * 4;
    u16*   hbuf      = (u16*)p;               // TOTAL_ROWS * IDIM * 2 bytes

    const int nOut = T_TOK * DMODEL;
    moe_zero_kernel<<<(nOut + 255) / 256, 256, 0, stream>>>(out, nOut, counts);
    moe_gate_kernel<<<T_TOK, 256, 0, stream>>>(x, gw, scores);
    moe_route_kernel<<<T_TOK / 256, 256, 0, stream>>>(scores, e_bias, counts, staged_id, staged_wt);
    moe_scan_kernel<<<1, 32, 0, stream>>>(counts, offsets);
    moe_ffn1_kernel<<<dim3(8, IDIM / 64, NEXP + 1), 256, 0, stream>>>(
        x, w1, w3, sw1, sw3, counts, offsets, staged_id, hbuf);
    moe_ffn2_kernel<<<dim3(8, DMODEL / 64, NEXP + 1), 256, 0, stream>>>(
        hbuf, w2, sw2, counts, offsets, staged_id, staged_wt, out);
}
